// PointNet2_32220844655235
// MI455X (gfx1250) — compile-verified, matrix operand order checked by reading, not run on hardware
//
#include <hip/hip_runtime.h>
#include <hip/hip_bf16.h>
#include <stdint.h>

// ---------------------------------------------------------------------------
// PointNet++ forward for MI455X (gfx1250, wave32, WMMA)
// Heavy stage (fp1: 262144 x (256->256) GEMM, ~34 GFLOP) runs on
// v_wmma_f32_16x16x32_bf16. Everything else is tiny (<=16 centroids) scalar.
// ---------------------------------------------------------------------------

#define BN 16
#define NP 16384

typedef __attribute__((ext_vector_type(16))) __bf16 v16bf;
typedef __attribute__((ext_vector_type(8)))  float  v8f;

static __device__ __forceinline__ unsigned short f2bf(float f) {
  union { float f; unsigned u; } x; x.f = f;
  unsigned u = x.u;
  unsigned r = u + 0x7FFFu + ((u >> 16) & 1u);   // round-to-nearest-even
  return (unsigned short)(r >> 16);
}

// --------------------------- l0 de-stride ----------------------------------
__global__ void k_extract(const float* __restrict__ xyz,
                          float* __restrict__ l0x, float* __restrict__ l0f) {
  int t = blockIdx.x * blockDim.x + threadIdx.x;        // B*NP threads
  int b = t >> 14, n = t & (NP - 1);
  size_t base = (size_t)b * 6 * NP;
  l0x[(size_t)t * 3 + 0] = xyz[base + 0 * NP + n];
  l0x[(size_t)t * 3 + 1] = xyz[base + 1 * NP + n];
  l0x[(size_t)t * 3 + 2] = xyz[base + 2 * NP + n];
  l0f[(size_t)t * 3 + 0] = xyz[base + 3 * NP + n];
  l0f[(size_t)t * 3 + 1] = xyz[base + 4 * NP + n];
  l0f[(size_t)t * 3 + 2] = xyz[base + 5 * NP + n];
}

// --------------------------- FPS over 16384 pts ----------------------------
__global__ __launch_bounds__(1024) void k_fps1(const float* __restrict__ l0x,
                                               float* __restrict__ out_xyz) {
  const int b = blockIdx.x, t = threadIdx.x;
  __shared__ float rv[1024];
  __shared__ int   ri[1024];
  __shared__ float cc[3];
  float ld[16];
#pragma unroll
  for (int i = 0; i < 16; ++i) ld[i] = 1e10f;
  if (t == 0) {
    const float* q = &l0x[(size_t)b * NP * 3];
    cc[0] = q[0]; cc[1] = q[1]; cc[2] = q[2];
  }
  __syncthreads();
  for (int s = 0; s < 16; ++s) {
    if (t == 0) {
      out_xyz[(b * 16 + s) * 3 + 0] = cc[0];
      out_xyz[(b * 16 + s) * 3 + 1] = cc[1];
      out_xyz[(b * 16 + s) * 3 + 2] = cc[2];
    }
    float cx = cc[0], cy = cc[1], cz = cc[2];
    float best = -1.0f; int bi = 0;
#pragma unroll
    for (int i = 0; i < 16; ++i) {
      int p = i * 1024 + t;
      const float* q = &l0x[((size_t)b * NP + p) * 3];
      float dx = q[0] - cx, dy = q[1] - cy, dz = q[2] - cz;
      float d = dx * dx + dy * dy + dz * dz;
      ld[i] = fminf(ld[i], d);
      if (ld[i] > best) { best = ld[i]; bi = p; }
    }
    rv[t] = best; ri[t] = bi;
    __syncthreads();
    for (int str = 512; str > 0; str >>= 1) {
      if (t < str) {
        float ov = rv[t + str]; int oi = ri[t + str];
        if (ov > rv[t] || (ov == rv[t] && oi < ri[t])) { rv[t] = ov; ri[t] = oi; }
      }
      __syncthreads();
    }
    if (t == 0) {
      const float* q = &l0x[((size_t)b * NP + ri[0]) * 3];
      cc[0] = q[0]; cc[1] = q[1]; cc[2] = q[2];
    }
    __syncthreads();
  }
}

// --------------------- ball-query over 16384 (wave/centroid) ---------------
__global__ __launch_bounds__(256) void k_bq1(const float* __restrict__ l0x,
                                             const float* __restrict__ nx,
                                             int* __restrict__ gidx) {
  int wave = threadIdx.x >> 5, lane = threadIdx.x & 31;
  int g = blockIdx.x * 8 + wave;           // 0..255 centroids
  int b = g >> 4;
  float cx = nx[g * 3 + 0], cy = nx[g * 3 + 1], cz = nx[g * 3 + 2];
  int found[16]; int cnt = 0;
  for (int base = 0; base < NP; base += 32) {
    int p = base + lane;
    const float* q = &l0x[((size_t)b * NP + p) * 3];
    float dx = q[0] - cx, dy = q[1] - cy, dz = q[2] - cz;
    bool in = (dx * dx + dy * dy + dz * dz) < 0.04f;
    unsigned m = (unsigned)__ballot(in);
    while (m && cnt < 16) {
      int j = __builtin_ctz(m);
      m &= m - 1;
      found[cnt++] = base + j;
    }
    if (cnt >= 16) break;
  }
  int f0 = found[0];
  for (int k = cnt; k < 16; ++k) found[k] = f0;
  if (lane < 16) gidx[g * 16 + lane] = found[lane];
}

// --------------------------- small FPS (<=16 pts) --------------------------
__global__ void k_fps_small(const float* __restrict__ src, int Nsrc, int npoint,
                            float* __restrict__ out) {
  int b = blockIdx.x, t = threadIdx.x;     // 32 threads
  __shared__ float dist[16];
  __shared__ float cc[3];
  __shared__ int sfar;
  if (t < Nsrc) dist[t] = 1e10f;
  if (t == 0) sfar = 0;
  __syncthreads();
  for (int s = 0; s < npoint; ++s) {
    if (t == 0) {
      const float* q = &src[(b * Nsrc + sfar) * 3];
      cc[0] = q[0]; cc[1] = q[1]; cc[2] = q[2];
      out[(b * npoint + s) * 3 + 0] = q[0];
      out[(b * npoint + s) * 3 + 1] = q[1];
      out[(b * npoint + s) * 3 + 2] = q[2];
    }
    __syncthreads();
    if (t < Nsrc) {
      float dx = src[(b * Nsrc + t) * 3 + 0] - cc[0];
      float dy = src[(b * Nsrc + t) * 3 + 1] - cc[1];
      float dz = src[(b * Nsrc + t) * 3 + 2] - cc[2];
      dist[t] = fminf(dist[t], dx * dx + dy * dy + dz * dz);
    }
    __syncthreads();
    if (t == 0) {
      float best = -1.0f; int bi = 0;
      for (int j = 0; j < Nsrc; ++j)
        if (dist[j] > best) { best = dist[j]; bi = j; }
      sfar = bi;
    }
    __syncthreads();
  }
}

// --------------------------- small ball query ------------------------------
__global__ void k_bq_small(const float* __restrict__ centers,
                           const float* __restrict__ src,
                           int S, int Nsrc, float r2, int* __restrict__ gidx) {
  int t = blockIdx.x * blockDim.x + threadIdx.x;
  if (t >= BN * S) return;
  int b = t / S;
  float cx = centers[t * 3 + 0], cy = centers[t * 3 + 1], cz = centers[t * 3 + 2];
  int cnt = 0, f0 = 0;
  for (int j = 0; j < Nsrc && cnt < 16; ++j) {
    float dx = src[(b * Nsrc + j) * 3 + 0] - cx;
    float dy = src[(b * Nsrc + j) * 3 + 1] - cy;
    float dz = src[(b * Nsrc + j) * 3 + 2] - cz;
    if (dx * dx + dy * dy + dz * dz < r2) {
      if (cnt == 0) f0 = j;
      gidx[t * 16 + cnt] = j;
      cnt++;
    }
  }
  for (int k = cnt; k < 16; ++k) gidx[t * 16 + k] = f0;
}

// ------------------- generic SA grouped MLP (3 layers + max) ---------------
__global__ __launch_bounds__(256) void k_sa_mlp(
    const float* __restrict__ src_xyz, const float* __restrict__ src_feat,
    int Cfeat, int Nsrc, int S,
    const float* __restrict__ new_xyz, const int* __restrict__ gidx,
    const float* __restrict__ W0, const float* __restrict__ b0, int C0,
    const float* __restrict__ W1, const float* __restrict__ b1, int C1,
    const float* __restrict__ W2, const float* __restrict__ b2, int C2,
    float* __restrict__ out) {
  __shared__ float g[16 * 260];
  __shared__ float h[16 * 256];
  const int bs = blockIdx.x, b = bs / S;
  const int tid = threadIdx.x;
  const int Cin = 3 + Cfeat;
  for (int k = 0; k < 16; ++k) {
    int j = gidx[bs * 16 + k];
    for (int c = tid; c < Cin; c += blockDim.x) {
      float v;
      if (c < 3) v = src_xyz[((size_t)b * Nsrc + j) * 3 + c] - new_xyz[bs * 3 + c];
      else       v = src_feat[((size_t)b * Nsrc + j) * Cfeat + (c - 3)];
      g[k * 260 + c] = v;
    }
  }
  __syncthreads();
  for (int o = tid; o < C0; o += blockDim.x) {
    for (int k = 0; k < 16; ++k) {
      float acc = b0[o];
      for (int c = 0; c < Cin; ++c) acc += W0[o * Cin + c] * g[k * 260 + c];
      h[k * 256 + o] = fmaxf(acc, 0.0f);
    }
  }
  __syncthreads();
  for (int o = tid; o < C1; o += blockDim.x) {
    for (int k = 0; k < 16; ++k) {
      float acc = b1[o];
      for (int c = 0; c < C0; ++c) acc += W1[o * C0 + c] * h[k * 256 + c];
      g[k * 260 + o] = fmaxf(acc, 0.0f);
    }
  }
  __syncthreads();
  for (int o = tid; o < C2; o += blockDim.x) {
    float m = -1e30f;
    for (int k = 0; k < 16; ++k) {
      float acc = b2[o];
      for (int c = 0; c < C1; ++c) acc += W2[o * C1 + c] * g[k * 260 + c];
      m = fmaxf(m, fmaxf(acc, 0.0f));
    }
    out[(size_t)bs * C2 + o] = m;
  }
}

// ------------------- generic FP (3-NN interp + 2-layer MLP) ----------------
__global__ __launch_bounds__(256) void k_fp(
    const float* __restrict__ unk_xyz, const float* __restrict__ kn_xyz,
    int Sk, int Su,
    const float* __restrict__ unk_feat, int Cu,
    const float* __restrict__ kn_feat, int Ck,
    const float* __restrict__ W0, const float* __restrict__ b0, int C0,
    const float* __restrict__ W1, const float* __restrict__ b1, int C1,
    float* __restrict__ out) {
  __shared__ float f[772];
  __shared__ float h[512];
  __shared__ int   sidx[3];
  __shared__ float sw[3];
  int bs = blockIdx.x, b = bs / Su;
  int tid = threadIdx.x;
  if (tid == 0) {
    float ux = unk_xyz[bs * 3 + 0], uy = unk_xyz[bs * 3 + 1], uz = unk_xyz[bs * 3 + 2];
    float d0 = 1e30f, d1 = 1e30f, d2 = 1e30f; int i0 = 0, i1 = 0, i2 = 0;
    for (int j = 0; j < Sk; ++j) {
      float dx = ux - kn_xyz[(b * Sk + j) * 3 + 0];
      float dy = uy - kn_xyz[(b * Sk + j) * 3 + 1];
      float dz = uz - kn_xyz[(b * Sk + j) * 3 + 2];
      float d = dx * dx + dy * dy + dz * dz;
      if (d < d0)      { d2 = d1; i2 = i1; d1 = d0; i1 = i0; d0 = d; i0 = j; }
      else if (d < d1) { d2 = d1; i2 = i1; d1 = d; i1 = j; }
      else if (d < d2) { d2 = d; i2 = j; }
    }
    float w0 = 1.0f / (d0 + 1e-8f), w1 = 1.0f / (d1 + 1e-8f), w2 = 1.0f / (d2 + 1e-8f);
    float ws = w0 + w1 + w2;
    sidx[0] = i0; sidx[1] = i1; sidx[2] = i2;
    sw[0] = w0 / ws; sw[1] = w1 / ws; sw[2] = w2 / ws;
  }
  __syncthreads();
  for (int c = tid; c < Ck; c += blockDim.x)
    f[c] = sw[0] * kn_feat[((size_t)b * Sk + sidx[0]) * Ck + c] +
           sw[1] * kn_feat[((size_t)b * Sk + sidx[1]) * Ck + c] +
           sw[2] * kn_feat[((size_t)b * Sk + sidx[2]) * Ck + c];
  for (int c = tid; c < Cu; c += blockDim.x)
    f[Ck + c] = unk_feat[(size_t)bs * Cu + c];
  __syncthreads();
  int Cin = Ck + Cu;
  for (int o = tid; o < C0; o += blockDim.x) {
    float acc = b0[o];
    for (int c = 0; c < Cin; ++c) acc += W0[(size_t)o * Cin + c] * f[c];
    h[o] = fmaxf(acc, 0.0f);
  }
  __syncthreads();
  for (int o = tid; o < C1; o += blockDim.x) {
    float acc = b1[o];
    for (int c = 0; c < C0; ++c) acc += W1[(size_t)o * C0 + c] * h[c];
    out[(size_t)bs * C1 + o] = fmaxf(acc, 0.0f);
  }
}

// ---------------- fp1 weight pack: f32 (256x256) -> bf16 WMMA B layout -----
// B operand (16-bit, 32x16): lane<16 holds K {0..7,16..23}, lane>=16 holds
// K {8..15,24..31}; VGPR v packs K pair (kb, kb+1); col = lane & 15.
__global__ void k_wpack(const float* __restrict__ W, unsigned* __restrict__ wp) {
  int t = blockIdx.x * 256 + threadIdx.x;   // 4096 = 16 ct * 8 ks * 32 lanes
  int ct = t >> 8, lane = t & 31;
  int ks = (t >> 5) & 7;
  int col = lane & 15, hi = lane >> 4;
#pragma unroll
  for (int v = 0; v < 8; ++v) {
    int kb = ((v < 4) ? 0 : 16) + hi * 8 + (v & 3) * 2;
    int K0 = ks * 32 + kb;
    unsigned short lo = f2bf(W[(ct * 16 + col) * 256 + K0]);
    unsigned short hh = f2bf(W[(ct * 16 + col) * 256 + K0 + 1]);
    wp[(size_t)t * 8 + v] = (unsigned)lo | ((unsigned)hh << 16);
  }
}

// ---------------- fp1 fused: 3-NN interp -> bf16 WMMA GEMM -> out ----------
__global__ __launch_bounds__(256) void k_fp1_wmma(
    const float* __restrict__ l0x, const float* __restrict__ l1xyz,
    const float* __restrict__ l1f, const unsigned* __restrict__ wp,
    const float* __restrict__ bias, float* __restrict__ out) {
  __shared__ float s_l1x[16 * 3];
  __shared__ float s_f[16 * 256];
  __shared__ unsigned short s_A[16 * 256];
  __shared__ int   s_i[16 * 3];
  __shared__ float s_w[16 * 3];
  const int tid = threadIdx.x;
  const int b  = blockIdx.x >> 10;          // 1024 row-tiles per batch
  const int n0 = (blockIdx.x & 1023) << 4;  // 16 points per tile
  if (tid < 48) s_l1x[tid] = l1xyz[b * 48 + tid];
  for (int i = tid; i < 16 * 256; i += 256) s_f[i] = l1f[(size_t)b * 16 * 256 + i];
  __syncthreads();
  if (tid < 16) {
    int n = n0 + tid;
    const float* q = &l0x[((size_t)b * NP + n) * 3];
    float ux = q[0], uy = q[1], uz = q[2];
    float d0 = 1e30f, d1 = 1e30f, d2 = 1e30f; int i0 = 0, i1 = 0, i2 = 0;
    for (int j = 0; j < 16; ++j) {
      float dx = ux - s_l1x[j * 3 + 0];
      float dy = uy - s_l1x[j * 3 + 1];
      float dz = uz - s_l1x[j * 3 + 2];
      float d = dx * dx + dy * dy + dz * dz;
      if (d < d0)      { d2 = d1; i2 = i1; d1 = d0; i1 = i0; d0 = d; i0 = j; }
      else if (d < d1) { d2 = d1; i2 = i1; d1 = d; i1 = j; }
      else if (d < d2) { d2 = d; i2 = j; }
    }
    float w0 = 1.0f / (d0 + 1e-8f), w1 = 1.0f / (d1 + 1e-8f), w2 = 1.0f / (d2 + 1e-8f);
    float ws = w0 + w1 + w2;
    s_i[tid * 3 + 0] = i0; s_i[tid * 3 + 1] = i1; s_i[tid * 3 + 2] = i2;
    s_w[tid * 3 + 0] = w0 / ws; s_w[tid * 3 + 1] = w1 / ws; s_w[tid * 3 + 2] = w2 / ws;
  }
  __syncthreads();
  // Build bf16 A tile (16 rows x 256 K) in LDS
  for (int r = 0; r < 16; ++r) {
    int i0 = s_i[r * 3 + 0], i1 = s_i[r * 3 + 1], i2 = s_i[r * 3 + 2];
    float w0 = s_w[r * 3 + 0], w1 = s_w[r * 3 + 1], w2 = s_w[r * 3 + 2];
    int k = tid;
    float v = w0 * s_f[i0 * 256 + k] + w1 * s_f[i1 * 256 + k] + w2 * s_f[i2 * 256 + k];
    s_A[r * 256 + k] = f2bf(v);
  }
  __syncthreads();
  const int wave = tid >> 5, lane = tid & 31;
  const int M = lane & 15, hi = lane >> 4;
  int kb[8];
#pragma unroll
  for (int v = 0; v < 8; ++v) kb[v] = ((v < 4) ? 0 : 16) + hi * 8 + (v & 3) * 2;
#pragma unroll
  for (int cti = 0; cti < 2; ++cti) {
    int ct = wave * 2 + cti;
    __builtin_prefetch(wp + (((size_t)ct * 8) * 32 + lane) * 8, 0, 0);
    union { v8f v; float f[8]; } acc;
#pragma unroll
    for (int r = 0; r < 8; ++r) acc.f[r] = 0.0f;
#pragma unroll
    for (int ks = 0; ks < 8; ++ks) {
      union { v16bf v; unsigned u[8]; } a;
#pragma unroll
      for (int v2 = 0; v2 < 8; ++v2) {
        int k0 = ks * 32 + kb[v2];
        a.u[v2] = (unsigned)s_A[M * 256 + k0] | ((unsigned)s_A[M * 256 + k0 + 1] << 16);
      }
      union { v16bf v; unsigned u[8]; } bm;
      const unsigned* wpp = wp + ((((size_t)ct * 8 + ks) * 32 + lane) << 3);
#pragma unroll
      for (int v2 = 0; v2 < 8; ++v2) bm.u[v2] = wpp[v2];
      acc.v = __builtin_amdgcn_wmma_f32_16x16x32_bf16(
          false, a.v, false, bm.v, (short)0, acc.v, false, false);
    }
    // C/D layout: VGPR r -> row (r + 8*hi), col = lane&15
#pragma unroll
    for (int r = 0; r < 8; ++r) {
      int row = r + hi * 8;
      int co = ct * 16 + M;
      float val = fmaxf(acc.f[r] + bias[co], 0.0f);
      out[((size_t)b * 256 + co) * NP + (n0 + row)] = val;
    }
  }
}

// ---------------------------------------------------------------------------
extern "C" void kernel_launch(void* const* d_in, const int* in_sizes, int n_in,
                              void* d_out, int out_size, void* d_ws, size_t ws_size,
                              hipStream_t stream) {
  const float* xyz    = (const float*)d_in[0];
  const float* sa1_w0 = (const float*)d_in[1];  const float* sa1_b0 = (const float*)d_in[2];
  const float* sa1_w1 = (const float*)d_in[3];  const float* sa1_b1 = (const float*)d_in[4];
  const float* sa1_w2 = (const float*)d_in[5];  const float* sa1_b2 = (const float*)d_in[6];
  const float* sa2_w0 = (const float*)d_in[7];  const float* sa2_b0 = (const float*)d_in[8];
  const float* sa2_w1 = (const float*)d_in[9];  const float* sa2_b1 = (const float*)d_in[10];
  const float* sa2_w2 = (const float*)d_in[11]; const float* sa2_b2 = (const float*)d_in[12];
  const float* sa3_w0 = (const float*)d_in[13]; const float* sa3_b0 = (const float*)d_in[14];
  const float* sa3_w1 = (const float*)d_in[15]; const float* sa3_b1 = (const float*)d_in[16];
  const float* sa3_w2 = (const float*)d_in[17]; const float* sa3_b2 = (const float*)d_in[18];
  const float* fp3_w0 = (const float*)d_in[19]; const float* fp3_b0 = (const float*)d_in[20];
  const float* fp3_w1 = (const float*)d_in[21]; const float* fp3_b1 = (const float*)d_in[22];
  const float* fp2_w0 = (const float*)d_in[23]; const float* fp2_b0 = (const float*)d_in[24];
  const float* fp2_w1 = (const float*)d_in[25]; const float* fp2_b1 = (const float*)d_in[26];
  const float* fp1_w0 = (const float*)d_in[27]; const float* fp1_b0 = (const float*)d_in[28];
  float* out = (float*)d_out;

  char* ws = (char*)d_ws;
  size_t off = 0;
  auto alloc = [&](size_t bytes) {
    off = (off + 255) & ~(size_t)255;
    void* p = ws + off;
    off += bytes;
    return p;
  };
  float* l0x     = (float*)alloc((size_t)BN * NP * 3 * 4);
  float* l0f     = (float*)alloc((size_t)BN * NP * 3 * 4);
  float* l1_xyz  = (float*)alloc(BN * 16 * 3 * 4);
  int*   l1_gidx = (int*)  alloc(BN * 16 * 16 * 4);
  float* l1_f    = (float*)alloc(BN * 16 * 128 * 4);
  float* l2_xyz  = (float*)alloc(BN * 12 * 3 * 4);
  int*   l2_gidx = (int*)  alloc(BN * 12 * 16 * 4);
  float* l2_f    = (float*)alloc(BN * 12 * 256 * 4);
  float* l3_xyz  = (float*)alloc(BN * 8 * 3 * 4);
  int*   l3_gidx = (int*)  alloc(BN * 8 * 16 * 4);
  float* l3_f    = (float*)alloc(BN * 8 * 512 * 4);
  float* l2_f2   = (float*)alloc(BN * 12 * 512 * 4);
  float* l1_f2   = (float*)alloc(BN * 16 * 256 * 4);
  unsigned* wpk  = (unsigned*)alloc(16 * 8 * 32 * 8 * 4);
  (void)ws_size; (void)in_sizes; (void)n_in; (void)out_size;

  k_extract<<<(BN * NP) / 256, 256, 0, stream>>>(xyz, l0x, l0f);
  k_fps1<<<BN, 1024, 0, stream>>>(l0x, l1_xyz);
  k_bq1<<<32, 256, 0, stream>>>(l0x, l1_xyz, l1_gidx);
  k_sa_mlp<<<BN * 16, 256, 0, stream>>>(l0x, l0f, 3, NP, 16, l1_xyz, l1_gidx,
                                        sa1_w0, sa1_b0, 32, sa1_w1, sa1_b1, 32,
                                        sa1_w2, sa1_b2, 128, l1_f);
  k_fps_small<<<BN, 32, 0, stream>>>(l1_xyz, 16, 12, l2_xyz);
  k_bq_small<<<3, 64, 0, stream>>>(l2_xyz, l1_xyz, 12, 16, 0.16f, l2_gidx);
  k_sa_mlp<<<BN * 12, 256, 0, stream>>>(l1_xyz, l1_f, 128, 16, 12, l2_xyz, l2_gidx,
                                        sa2_w0, sa2_b0, 128, sa2_w1, sa2_b1, 128,
                                        sa2_w2, sa2_b2, 256, l2_f);
  k_fps_small<<<BN, 32, 0, stream>>>(l2_xyz, 12, 8, l3_xyz);
  k_bq_small<<<2, 64, 0, stream>>>(l3_xyz, l2_xyz, 8, 12, 0.64f, l3_gidx);
  k_sa_mlp<<<BN * 8, 256, 0, stream>>>(l2_xyz, l2_f, 256, 12, 8, l3_xyz, l3_gidx,
                                       sa3_w0, sa3_b0, 256, sa3_w1, sa3_b1, 256,
                                       sa3_w2, sa3_b2, 512, l3_f);
  k_fp<<<BN * 12, 256, 0, stream>>>(l2_xyz, l3_xyz, 8, 12, l2_f, 256, l3_f, 512,
                                    fp3_w0, fp3_b0, 512, fp3_w1, fp3_b1, 512, l2_f2);
  k_fp<<<BN * 16, 256, 0, stream>>>(l1_xyz, l2_xyz, 12, 16, l1_f, 128, l2_f2, 512,
                                    fp2_w0, fp2_b0, 256, fp2_w1, fp2_b1, 256, l1_f2);
  k_wpack<<<16, 256, 0, stream>>>(fp1_w0, wpk);
  k_fp1_wmma<<<(BN * NP) / 16, 256, 0, stream>>>(l0x, l1_xyz, l1_f2, wpk, fp1_b0, out);
}
